// RealImagFourValLinear_73555609911440
// MI455X (gfx1250) — matrix operand order repaired and checked by reading, not compiled, not run on hardware
//
#include <hip/hip_runtime.h>
#include <hip/hip_bf16.h>

// ---------------------------------------------------------------------------
// RealImagFourValLinear for MI455X (gfx1250, wave32, WMMA)
//
// W1 = bf16(s_re * Q_re), W2 = bf16(s_im * Q_im),  Q in {+-0.5, +-1}
//   y_re = X_re @ W1^T - X_im @ W2^T
//   y_im = X_re @ W2^T + X_im @ W1^T
// Core op: v_wmma_f32_16x16x32_bf16. Block tile 128x128, wave tile 32x64,
// BK=32, register-prefetch software pipeline over a single LDS buffer.
// ---------------------------------------------------------------------------

typedef __attribute__((ext_vector_type(16))) __bf16        v16bf;
typedef __attribute__((ext_vector_type(8)))  __bf16        v8bf;
typedef __attribute__((ext_vector_type(8)))  float         v8f;
typedef __attribute__((ext_vector_type(8)))  unsigned int  v8u;

#define N_ROWS 8192
#define K_DIM  2048
#define O_DIM  2048

#define BM 128   // block rows of X
#define BN 128   // block cols (output features)
#define BK 32    // K per stage == WMMA K
#define LDS_PITCH 40  // padded LDS row (ushorts): 80B rows, 16B-aligned chunks

// f32 -> bf16, round-to-nearest-even, via bits (no codegen dependence).
static __device__ __forceinline__ unsigned short f2bf(float f) {
  unsigned int u = __float_as_uint(f);
  unsigned int lsb = (u >> 16) & 1u;
  u += 0x7fffu + lsb;
  return (unsigned short)(u >> 16);
}

// Build a 16-element bf16 fragment from two 8-element 16B-aligned chunks.
static __device__ __forceinline__ v16bf frag16(const unsigned short* p0,
                                               const unsigned short* p1) {
  v8bf a = *(const v8bf*)p0;
  v8bf b = *(const v8bf*)p1;
  v16bf r;
#pragma unroll
  for (int i = 0; i < 8; ++i) { r[i] = a[i]; r[i + 8] = b[i]; }
  return r;
}

// Negate all 16 bf16 lanes-elements (sign-bit XOR; bf16 A/B NEG modifiers are
// illegal on WMMA per ISA, so negation must be done in VALU).
static __device__ __forceinline__ v16bf negbf(v16bf x) {
  v8u u;
  __builtin_memcpy(&u, &x, 32);
  u ^= 0x80008000u;
  v16bf r;
  __builtin_memcpy(&r, &u, 32);
  return r;
}

// --------------------------- Phase 1: |W| partial sums ----------------------
__global__ __launch_bounds__(256) void abs_sum_partial(
    const float4* __restrict__ wre, const float4* __restrict__ wim,
    float* __restrict__ part, int n4) {
  __shared__ float red[256];
  const int t = threadIdx.x;
  float sre = 0.f, sim = 0.f;
  for (int i = blockIdx.x * 256 + t; i < n4; i += 256 * gridDim.x) {
    float4 a = wre[i];
    sre += fabsf(a.x) + fabsf(a.y) + fabsf(a.z) + fabsf(a.w);
    float4 b = wim[i];
    sim += fabsf(b.x) + fabsf(b.y) + fabsf(b.z) + fabsf(b.w);
  }
  red[t] = sre; __syncthreads();
  for (int s = 128; s > 0; s >>= 1) {
    if (t < s) red[t] += red[t + s];
    __syncthreads();
  }
  if (t == 0) part[blockIdx.x] = red[0];
  __syncthreads();
  red[t] = sim; __syncthreads();
  for (int s = 128; s > 0; s >>= 1) {
    if (t < s) red[t] += red[t + s];
    __syncthreads();
  }
  if (t == 0) part[256 + blockIdx.x] = red[0];
}

// --------------------------- Phase 2: finalize scales -----------------------
__global__ __launch_bounds__(256) void finalize_scales(
    const float* __restrict__ part, float* __restrict__ scales) {
  __shared__ float red[256];
  const int t = threadIdx.x;
  const float inv_n = 1.0f / (2048.0f * 2048.0f);
  red[t] = part[t]; __syncthreads();
  for (int s = 128; s > 0; s >>= 1) {
    if (t < s) red[t] += red[t + s];
    __syncthreads();
  }
  if (t == 0) scales[0] = fmaxf(red[0] * inv_n, 1e-6f);
  __syncthreads();
  red[t] = part[256 + t]; __syncthreads();
  for (int s = 128; s > 0; s >>= 1) {
    if (t < s) red[t] += red[t + s];
    __syncthreads();
  }
  if (t == 0) scales[1] = fmaxf(red[0] * inv_n, 1e-6f);
}

// ------------------- Phase 3: quantize + fold scale into bf16 ---------------
static __device__ __forceinline__ float quant4(float w, float s) {
  float ws = w / s;
  float wq = rintf(ws * 2.0f) * 0.5f;          // jnp.round = RNE
  if (fabsf(wq) < 1e-5f) wq = (ws >= 0.f) ? 0.5f : -0.5f;
  return fminf(1.0f, fmaxf(-1.0f, wq));        // in {+-0.5, +-1}
}

__global__ __launch_bounds__(256) void quantize_kernel(
    const float* __restrict__ wre, const float* __restrict__ wim,
    const float* __restrict__ scales,
    unsigned short* __restrict__ w1, unsigned short* __restrict__ w2, int n) {
  const float sre = scales[0], sim = scales[1];
  for (int i = blockIdx.x * blockDim.x + threadIdx.x; i < n;
       i += blockDim.x * gridDim.x) {
    w1[i] = f2bf(sre * quant4(wre[i], sre));   // s_re * Q_re
    w2[i] = f2bf(sim * quant4(wim[i], sim));   // s_im * Q_im
  }
}

// --------------------------- Phase 4: X f32 -> bf16 -------------------------
__global__ __launch_bounds__(256) void convert_x(
    const float4* __restrict__ xre, const float4* __restrict__ xim,
    ushort4* __restrict__ xre16, ushort4* __restrict__ xim16, int n4) {
  for (int i = blockIdx.x * blockDim.x + threadIdx.x; i < n4;
       i += blockDim.x * gridDim.x) {
    float4 a = xre[i];
    xre16[i] = make_ushort4(f2bf(a.x), f2bf(a.y), f2bf(a.z), f2bf(a.w));
    float4 b = xim[i];
    xim16[i] = make_ushort4(f2bf(b.x), f2bf(b.y), f2bf(b.z), f2bf(b.w));
  }
}

// --------------------------- Phase 5: WMMA GEMM -----------------------------
#define WMMA_BF16(A, B, C) \
  __builtin_amdgcn_wmma_f32_16x16x32_bf16(false, (A), false, (B), (short)0, (C), false, false)

__global__ __launch_bounds__(256) void gemm_kernel(
    const unsigned short* __restrict__ Xre, const unsigned short* __restrict__ Xim,
    const unsigned short* __restrict__ W1, const unsigned short* __restrict__ W2,
    float* __restrict__ Yre, float* __restrict__ Yim) {
  __shared__ unsigned short sXre[BM * LDS_PITCH];
  __shared__ unsigned short sXim[BM * LDS_PITCH];
  __shared__ unsigned short sW1 [BN * LDS_PITCH];
  __shared__ unsigned short sW2 [BN * LDS_PITCH];

  const int tid  = threadIdx.x;
  const int row0 = blockIdx.y * BM;   // X rows     (8192/128 = 64 blocks)
  const int col0 = blockIdx.x * BN;   // out feats  (2048/128 = 16 blocks)

  const int w    = tid >> 5;          // 8 waves: 4 (m) x 2 (n)
  const int lane = tid & 31;
  const int lm   = lane & 15;
  const int lh   = lane >> 4;
  const int wm   = (w >> 1) * 32;     // wave m-offset: 0/32/64/96
  const int wn   = (w & 1) * 64;      // wave n-offset: 0/64 (4 n-tiles each)

  // Staging indices: 128 rows x 4 chunks(8 bf16) per matrix = 512 x 16B.
  int sr[2], sc[2];
#pragma unroll
  for (int i = 0; i < 2; ++i) {
    int idx = tid + i * 256;
    sr[i] = idx >> 2;
    sc[i] = (idx & 3) * 8;
  }

  v8f accRe[2][4] = {};   // [tm][tn]
  v8f accIm[2][4] = {};

  uint4 pXre[2], pXim[2], pW1[2], pW2[2];

  // Prologue: prefetch K-slab 0 into registers.
#pragma unroll
  for (int i = 0; i < 2; ++i) {
    size_t gx = (size_t)(row0 + sr[i]) * K_DIM + sc[i];
    size_t gw = (size_t)(col0 + sr[i]) * K_DIM + sc[i];
    pXre[i] = *(const uint4*)(Xre + gx);
    pXim[i] = *(const uint4*)(Xim + gx);
    pW1[i]  = *(const uint4*)(W1 + gw);
    pW2[i]  = *(const uint4*)(W2 + gw);
  }

  for (int kb = 0; kb < K_DIM; kb += BK) {
    // ---- commit prefetched slab to LDS ----
#pragma unroll
    for (int i = 0; i < 2; ++i) {
      int o = sr[i] * LDS_PITCH + sc[i];
      *(uint4*)&sXre[o] = pXre[i];
      *(uint4*)&sXim[o] = pXim[i];
      *(uint4*)&sW1[o]  = pW1[i];
      *(uint4*)&sW2[o]  = pW2[i];
    }
    __syncthreads();

    // ---- issue next slab's global loads early (latency hidden by WMMAs) ----
    if (kb + BK < K_DIM) {
#pragma unroll
      for (int i = 0; i < 2; ++i) {
        size_t gx = (size_t)(row0 + sr[i]) * K_DIM + (kb + BK) + sc[i];
        size_t gw = (size_t)(col0 + sr[i]) * K_DIM + (kb + BK) + sc[i];
        pXre[i] = *(const uint4*)(Xre + gx);
        pXim[i] = *(const uint4*)(Xim + gx);
        pW1[i]  = *(const uint4*)(W1 + gw);
        pW2[i]  = *(const uint4*)(W2 + gw);
      }
    }

    // ---- A fragments (rows, contiguous-K chunks at lh*8 and lh*8+16) ----
    v16bf aRe[2], aIm[2], aImN[2];
#pragma unroll
    for (int tm = 0; tm < 2; ++tm) {
      int base = (wm + tm * 16 + lm) * LDS_PITCH + lh * 8;
      aRe[tm]  = frag16(&sXre[base], &sXre[base + 16]);
      aIm[tm]  = frag16(&sXim[base], &sXim[base + 16]);
      aImN[tm] = negbf(aIm[tm]);
    }
    // ---- B fragments (col o = lm, K chunks at lh*8 and lh*8+16), 32 WMMAs ----
#pragma unroll
    for (int tn = 0; tn < 4; ++tn) {
      int base = (wn + tn * 16 + lm) * LDS_PITCH + lh * 8;
      v16bf b1 = frag16(&sW1[base], &sW1[base + 16]);
      v16bf b2 = frag16(&sW2[base], &sW2[base + 16]);
#pragma unroll
      for (int tm = 0; tm < 2; ++tm) {
        accRe[tm][tn] = WMMA_BF16(aRe[tm],  b1, accRe[tm][tn]);
        accRe[tm][tn] = WMMA_BF16(aImN[tm], b2, accRe[tm][tn]);
        accIm[tm][tn] = WMMA_BF16(aRe[tm],  b2, accIm[tm][tn]);
        accIm[tm][tn] = WMMA_BF16(aIm[tm],  b1, accIm[tm][tn]);
      }
    }
    __syncthreads();
  }

  // ---- epilogue: scales already folded; pure stores ----
#pragma unroll
  for (int tm = 0; tm < 2; ++tm) {
#pragma unroll
    for (int tn = 0; tn < 4; ++tn) {
      int r = row0 + wm + tm * 16 + lh * 8;
      int c = col0 + wn + tn * 16 + lm;
#pragma unroll
      for (int v = 0; v < 8; ++v) {
        size_t o = (size_t)(r + v) * O_DIM + c;
        Yre[o] = accRe[tm][tn][v];
        Yim[o] = accIm[tm][tn][v];
      }
    }
  }
}

// ---------------------------------------------------------------------------
extern "C" void kernel_launch(void* const* d_in, const int* in_sizes, int n_in,
                              void* d_out, int out_size, void* d_ws, size_t ws_size,
                              hipStream_t stream) {
  const float* xre = (const float*)d_in[0];
  const float* xim = (const float*)d_in[1];
  const float* wre = (const float*)d_in[2];
  const float* wim = (const float*)d_in[3];

  float* yre = (float*)d_out;
  float* yim = yre + (size_t)N_ROWS * O_DIM;

  // Workspace layout (~80.004 MiB total):
  //   [0, 2KiB)      : 512 partial sums
  //   [2KiB, +8B)    : scales (s_re, s_im)
  //   [4KiB, +8MiB)  : W1 = s_re*Q_re  bf16 (2048x2048)
  //   [...,  +8MiB)  : W2 = s_im*Q_im  bf16
  //   [..., +32MiB)  : X_re bf16 (8192x2048)
  //   [..., +32MiB)  : X_im bf16
  char* ws = (char*)d_ws;
  float* part           = (float*)ws;
  float* scales         = (float*)(ws + 2048);
  unsigned short* w1    = (unsigned short*)(ws + 4096);
  unsigned short* w2    = w1 + (size_t)O_DIM * K_DIM;
  unsigned short* xre16 = (unsigned short*)(ws + 4096 + (size_t)16 * 1024 * 1024);
  unsigned short* xim16 = xre16 + (size_t)N_ROWS * K_DIM;

  const int nW  = O_DIM * K_DIM;       // 4M weights per matrix
  const int nW4 = nW / 4;
  const int nX4 = N_ROWS * K_DIM / 4;  // 4M float4 per matrix

  abs_sum_partial<<<256, 256, 0, stream>>>(
      (const float4*)wre, (const float4*)wim, part, nW4);
  finalize_scales<<<1, 256, 0, stream>>>(part, scales);
  quantize_kernel<<<1024, 256, 0, stream>>>(wre, wim, scales, w1, w2, nW);
  convert_x<<<2048, 256, 0, stream>>>(
      (const float4*)xre, (const float4*)xim,
      (ushort4*)xre16, (ushort4*)xim16, nX4);

  dim3 grid(O_DIM / BN, N_ROWS / BM);  // (16, 64)
  gemm_kernel<<<grid, 256, 0, stream>>>(xre16, xim16, w1, w2, yre, yim);
}
